// PositionAttentionBlock_73478300500483
// MI455X (gfx1250) — matrix operand order repaired and checked by reading, not compiled, not run on hardware
//
#include <hip/hip_runtime.h>
#include <hip/hip_bf16.h>

typedef __attribute__((ext_vector_type(16))) _Float16 v16h;
typedef __attribute__((ext_vector_type(8)))  _Float16 v8h;
typedef __attribute__((ext_vector_type(8)))  float    v8f;

#define C2    64
#define CIN   128
#define NPIX  4096     // 64*64
#define BATCH 4
#define BN_EPS 1e-5f

union V16H { v16h v; float4 f4[2]; };

// ---------------------------------------------------------------------------
// Kernel 1: 1x1-conv projections -> f16 workspace
//   which==0: theta(HL) -> Kw  [b][n][64]   (row-major over channels)
//   which==1: phi(HL)   -> Qw  [b][n][64]
//   which==2: g(L)      -> Vw  [b][c][4096] (channel-major)
// ---------------------------------------------------------------------------
__global__ __launch_bounds__(256)
void pa_proj_kernel(const float* __restrict__ Hbuf, const float* __restrict__ Lbuf,
                    const float* __restrict__ thw, const float* __restrict__ thb,
                    const float* __restrict__ phw, const float* __restrict__ phb,
                    const float* __restrict__ gw,  const float* __restrict__ gb,
                    _Float16* __restrict__ Kw, _Float16* __restrict__ Qw,
                    _Float16* __restrict__ Vw)
{
    const int which = blockIdx.y;
    const int b     = blockIdx.x >> 4;                    // 16 blocks of 256 per image
    const int n     = ((blockIdx.x & 15) << 8) + threadIdx.x;

    const float* w;  const float* bias;  int cin;
    if (which == 0)      { w = thw; bias = thb; cin = CIN; }
    else if (which == 1) { w = phw; bias = phb; cin = CIN; }
    else                 { w = gw;  bias = gb;  cin = C2;  }

    float acc[C2];
#pragma unroll
    for (int c = 0; c < C2; ++c) acc[c] = 0.0f;

    for (int ci = 0; ci < cin; ++ci) {
        float x;
        if (which == 2)      x = Lbuf[((size_t)(b * C2 + ci)) * NPIX + n];
        else if (ci < C2)    x = Hbuf[((size_t)(b * C2 + ci)) * NPIX + n];
        else                 x = Lbuf[((size_t)(b * C2 + (ci - C2))) * NPIX + n];
        const float* wrow = w + ci;                       // uniform -> scalar loads
#pragma unroll
        for (int c = 0; c < C2; ++c) acc[c] += wrow[(size_t)c * cin] * x;
    }

    if (which == 0) {
        _Float16* o = Kw + ((size_t)b * NPIX + n) * C2;
#pragma unroll
        for (int c = 0; c < C2; ++c) o[c] = (_Float16)(acc[c] + bias[c]);
    } else if (which == 1) {
        _Float16* o = Qw + ((size_t)b * NPIX + n) * C2;
#pragma unroll
        for (int c = 0; c < C2; ++c) o[c] = (_Float16)(acc[c] + bias[c]);
    } else {
        _Float16* o = Vw + (size_t)b * C2 * NPIX + n;
#pragma unroll
        for (int c = 0; c < C2; ++c) o[(size_t)c * NPIX] = (_Float16)(acc[c] + bias[c]);
    }
}

// ---------------------------------------------------------------------------
// Kernel 2: attention core. One WG (256 thr = 8 waves) per (batch, 16-col block).
//   Phase A: S[m][n] = sum_c Qw[m][c]*Kw[n][c]  via v_wmma_f32_16x16x32_f16 -> LDS (f16)
//   Phase B: exact softmax over n (per m), vectorized b128 LDS, exp in place
//   Phase C: E[c][m] = (1/colsum) * sum_n Vw[c][n]*P[n][m]; all 8 waves
//            (4 c-tiles x 2 n-halves), partials combined through LDS scratch.
// Dynamic LDS: 16*4096 f16 stripe + reduction scratch + 4 partial f32 tiles
// ---------------------------------------------------------------------------
#define SMEM_BYTES (16 * NPIX * 2 + (256 + 16 + 16 + 4 * 256) * 4)   // 136320 B

__global__ __launch_bounds__(256)
void pa_attn_kernel(const _Float16* __restrict__ Kw, const _Float16* __restrict__ Qw,
                    const _Float16* __restrict__ Vw, float* __restrict__ Ebuf)
{
    extern __shared__ char smem[];
    _Float16* Sbuf    = (_Float16*)smem;                    // [16][4096] f16
    float*    scratch = (float*)(smem + 16 * NPIX * 2);     // [16][16]
    float*    cmax    = scratch + 256;                      // [16]
    float*    cinv    = cmax + 16;                          // [16]
    float*    ptile   = cinv + 16;                          // [4][256] partial tiles

    const int b    = blockIdx.x >> 8;
    const int m0   = (blockIdx.x & 255) << 4;
    const int tid  = threadIdx.x;
    const int lane = tid & 31;
    const int wave = tid >> 5;

    // ---- Phase A: score stripe -------------------------------------------
    // B-matrix (Q^T) tiles: col m = lane&15, k=c contiguous; load once per wave.
    V16H qb[2];
    {
        const _Float16* qbase = Qw + ((size_t)b * NPIX + m0) * C2;
        const _Float16* p0 = qbase + (size_t)(lane & 15) * C2 + ((lane >> 4) << 4);
        qb[0].f4[0] = *(const float4*)(p0);
        qb[0].f4[1] = *(const float4*)(p0 + 8);
        qb[1].f4[0] = *(const float4*)(p0 + 32);
        qb[1].f4[1] = *(const float4*)(p0 + 40);
    }
    const _Float16* kbase = Kw + (size_t)b * NPIX * C2;
    for (int t = wave; t < 256; t += 8) {                   // 256 tiles of 16 n
        const int n0 = t << 4;
        // prefetch next tile's K rows (stream, lowers to global_prefetch_b8)
        __builtin_prefetch(kbase + (size_t)(n0 + 128 + (lane & 15)) * C2, 0, 1);
        v8f acc = {};
#pragma unroll
        for (int cc = 0; cc < 2; ++cc) {
            V16H ka;   // A-matrix: row n = lane&15, K contiguous 16h..16h+15
            const _Float16* p = kbase + (size_t)(n0 + (lane & 15)) * C2
                                      + (cc << 5) + ((lane >> 4) << 4);
            ka.f4[0] = *(const float4*)(p);
            ka.f4[1] = *(const float4*)(p + 8);
            acc = __builtin_amdgcn_wmma_f32_16x16x32_f16(
                false, ka.v, false, qb[cc].v, (short)0, acc, false, false);
        }
        const int m  = lane & 15;
        const int nb = n0 + ((lane >> 4) << 3);
#pragma unroll
        for (int v = 0; v < 8; ++v)
            Sbuf[(size_t)m * NPIX + nb + v] = (_Float16)acc[v];
    }
    __syncthreads();

    // ---- Phase B: softmax over n (exact, stripe resident, b128 LDS) ------
    {
        const int m = tid >> 4, j = tid & 15;
        const v8h* row = (const v8h*)(Sbuf + (size_t)m * NPIX + j * 256);
        float lmax = -3.0e38f;
        for (int k = 0; k < 32; ++k) {                      // 32 x 8 halves
            v8h x = row[k];
#pragma unroll
            for (int i = 0; i < 8; ++i) lmax = fmaxf(lmax, (float)x[i]);
        }
        scratch[m * 16 + j] = lmax;
    }
    __syncthreads();
    if (tid < 16) {
        float mx = scratch[tid * 16];
#pragma unroll
        for (int j = 1; j < 16; ++j) mx = fmaxf(mx, scratch[tid * 16 + j]);
        cmax[tid] = mx;
    }
    __syncthreads();
    {
        const int m = tid >> 4, j = tid & 15;
        const float mx = cmax[m];
        v8h* row = (v8h*)(Sbuf + (size_t)m * NPIX + j * 256);
        float lsum = 0.0f;
        for (int k = 0; k < 32; ++k) {
            v8h x = row[k];
            v8h y;
#pragma unroll
            for (int i = 0; i < 8; ++i) {
                float e = __expf((float)x[i] - mx);
                y[i] = (_Float16)e;
                lsum += e;
            }
            row[k] = y;                                     // exp in place -> P
        }
        scratch[m * 16 + j] = lsum;
    }
    __syncthreads();
    if (tid < 16) {
        float s = 0.0f;
#pragma unroll
        for (int j = 0; j < 16; ++j) s += scratch[tid * 16 + j];
        cinv[tid] = 1.0f / s;
    }
    __syncthreads();

    // ---- Phase C: E tile = V @ P, 8 waves = 4 c-tiles x 2 n-halves -------
    {
        const int c0  = (wave & 3) << 4;
        const int nb2 = (wave >> 2) << 11;                  // 0 or 2048
        const _Float16* vbase = Vw + (size_t)b * C2 * NPIX
                                   + (size_t)(c0 + (lane & 15)) * NPIX + nb2;
        const _Float16* pbase = Sbuf + (size_t)(lane & 15) * NPIX + nb2;
        v8f acc = {};
        for (int it = 0; it < 64; ++it) {                   // n in chunks of 32
            const int n0 = it << 5;
            V16H va, pb;
            const _Float16* p = vbase + n0 + ((lane >> 4) << 4);
            __builtin_prefetch(p + 64, 0, 1);               // V chunk two iterations ahead
            va.f4[0] = *(const float4*)(p);
            va.f4[1] = *(const float4*)(p + 8);
            const _Float16* q = pbase + n0 + ((lane >> 4) << 4);
            pb.f4[0] = *(const float4*)(q);                 // ds_load_b128
            pb.f4[1] = *(const float4*)(q + 8);
            acc = __builtin_amdgcn_wmma_f32_16x16x32_f16(
                false, va.v, false, pb.v, (short)0, acc, false, false);
        }
        if (wave >= 4) {                                    // stash high-half partial
            float* dst = ptile + ((wave & 3) << 8);
#pragma unroll
            for (int v = 0; v < 8; ++v) dst[v * 32 + lane] = acc[v];
        }
        __syncthreads();
        if (wave < 4) {                                     // combine + normalize
            const float* src = ptile + (wave << 8);
#pragma unroll
            for (int v = 0; v < 8; ++v) acc[v] += src[v * 32 + lane];
            const float inv = cinv[lane & 15];
            const int   m   = m0 + (lane & 15);
            const int   cb  = c0 + ((lane >> 4) << 3);
#pragma unroll
            for (int v = 0; v < 8; ++v)
                Ebuf[((size_t)b * C2 + cb + v) * NPIX + m] = acc[v] * inv;
        }
    }
}

// ---------------------------------------------------------------------------
// Kernel 3: conv3x3(pad=1) + BatchNorm(inference) + ReLU + residual add
// ---------------------------------------------------------------------------
__global__ __launch_bounds__(256)
void pa_cbr_kernel(const float* __restrict__ E, const float* __restrict__ cw,
                   const float* __restrict__ cb, const float* __restrict__ bng,
                   const float* __restrict__ bnb, const float* __restrict__ bnm,
                   const float* __restrict__ bnv, const float* __restrict__ Hbuf,
                   float* __restrict__ out)
{
    const int idx = blockIdx.x * 256 + threadIdx.x;
    const int p = idx & (NPIX - 1);
    const int c = (idx >> 12) & (C2 - 1);
    const int b = idx >> 18;
    const int y = p >> 6, x = p & 63;

    float s = cb[c];
    for (int ci = 0; ci < C2; ++ci) {
        const float* ep = E + ((size_t)(b * C2 + ci) << 12);
        const float* wp = cw + (size_t)(c * C2 + ci) * 9;   // uniform -> scalar
#pragma unroll
        for (int ky = 0; ky < 3; ++ky) {
            const int yy = y + ky - 1;
            if ((unsigned)yy >= 64u) continue;
#pragma unroll
            for (int kx = 0; kx < 3; ++kx) {
                const int xx = x + kx - 1;
                if ((unsigned)xx >= 64u) continue;
                s += wp[ky * 3 + kx] * ep[yy * 64 + xx];
            }
        }
    }
    float r = (s - bnm[c]) * rsqrtf(bnv[c] + BN_EPS) * bng[c] + bnb[c];
    r = fmaxf(r, 0.0f);
    out[idx] = Hbuf[idx] + r;
}

// ---------------------------------------------------------------------------
extern "C" void kernel_launch(void* const* d_in, const int* in_sizes, int n_in,
                              void* d_out, int out_size, void* d_ws, size_t ws_size,
                              hipStream_t stream)
{
    const float* Hbuf = (const float*)d_in[0];
    const float* Lbuf = (const float*)d_in[1];
    const float* thw  = (const float*)d_in[2];
    const float* thb  = (const float*)d_in[3];
    const float* phw  = (const float*)d_in[4];
    const float* phb  = (const float*)d_in[5];
    const float* gw   = (const float*)d_in[6];
    const float* gb   = (const float*)d_in[7];
    const float* cw   = (const float*)d_in[8];
    const float* cb   = (const float*)d_in[9];
    const float* bng  = (const float*)d_in[10];
    const float* bnb  = (const float*)d_in[11];
    const float* bnm  = (const float*)d_in[12];
    const float* bnv  = (const float*)d_in[13];
    float* out = (float*)d_out;

    // Workspace layout (10 MB total):
    char* ws = (char*)d_ws;
    _Float16* Kw = (_Float16*)(ws);                 // 2 MB  [B][N][64]
    _Float16* Qw = (_Float16*)(ws + (2u << 20));    // 2 MB  [B][N][64]
    _Float16* Vw = (_Float16*)(ws + (4u << 20));    // 2 MB  [B][64][N]
    float*    Eb = (float*)   (ws + (6u << 20));    // 4 MB  [B][64][N]

    (void)hipFuncSetAttribute((const void*)pa_attn_kernel,
                              hipFuncAttributeMaxDynamicSharedMemorySize,
                              SMEM_BYTES);

    dim3 gproj(BATCH * (NPIX / 256), 3, 1);
    pa_proj_kernel<<<gproj, 256, 0, stream>>>(Hbuf, Lbuf, thw, thb, phw, phb,
                                              gw, gb, Kw, Qw, Vw);

    dim3 gattn(BATCH * (NPIX / 16), 1, 1);          // 1024 WGs
    pa_attn_kernel<<<gattn, 256, SMEM_BYTES, stream>>>(Kw, Qw, Vw, Eb);

    dim3 gcbr((BATCH * C2 * NPIX) / 256, 1, 1);     // 4096 WGs
    pa_cbr_kernel<<<gcbr, 256, 0, stream>>>(Eb, cw, cb, bng, bnb, bnm, bnv,
                                            Hbuf, out);
}
